// SpikeSelfAttentionBlock_86071144611850
// MI455X (gfx1250) — matrix operand order repaired and checked by reading, not compiled, not run on hardware
//
#include <hip/hip_runtime.h>

// Problem constants (reference: T=16, B=1024, E=1024, H=16, D=64)
constexpr int cT = 16;
constexpr int cB = 1024;
constexpr int cE = 1024;
constexpr int cH = 16;
constexpr int cD = 64;
constexpr int cTB = cT * cB;        // 16384 rows of the big GEMMs
constexpr long cBE = (long)cB * cE; // 1,048,576

typedef __bf16 bf16_t;
typedef __attribute__((ext_vector_type(16))) bf16_t v16bf;
typedef __attribute__((ext_vector_type(8)))  bf16_t v8bf;
typedef __attribute__((ext_vector_type(8)))  float  v8f;
typedef int v4i_t __attribute__((ext_vector_type(4)));
typedef __attribute__((address_space(1))) v4i_t gvec_t;  // global int4
typedef __attribute__((address_space(3))) v4i_t lvec_t;  // LDS int4

// Async global->LDS staging path (ASYNCcnt) if the toolchain exposes it.
#if defined(__has_builtin)
#if __has_builtin(__builtin_amdgcn_global_load_async_to_lds_b128)
#define HAVE_ASYNC_LDS 1
#endif
#endif
#ifndef HAVE_ASYNC_LDS
#define HAVE_ASYNC_LDS 0
#endif

__device__ __forceinline__ void cp16_g2l(const bf16_t* g, bf16_t* l) {
#if HAVE_ASYNC_LDS
  __builtin_amdgcn_global_load_async_to_lds_b128(
      (gvec_t*)(void*)g, (lvec_t*)(void*)l, 0, 0);
#else
  *(v8bf*)l = *(const v8bf*)g;
#endif
}

__device__ __forceinline__ void wait_stage() {
#if HAVE_ASYNC_LDS
  asm volatile("s_wait_asynccnt 0x0" ::: "memory");
#endif
}

// ---------------------------------------------------------------------------
// WMMA bf16 16x16x32 fragment layouts (wave32):
// A (16x32, MxK): lane L holds row M=L%16;
//   elems 0..7  -> K = (L/16)*8 + i ; elems 8..15 -> K = 16 + (L/16)*8 + i
// B (32x16, KxN): lane L holds column N=L%16;
//   elems 0..7  -> K = (L/16)*8 + i ; elems 8..15 -> K = 16 + (L/16)*8 + i (same as A)
// B here is W^T, so column n of B == row n of W (contiguous in K).
// ---------------------------------------------------------------------------
__device__ __forceinline__ v16bf loadA_g(const bf16_t* __restrict__ p, long ld,
                                         int row0, int kbase, int lane) {
  const bf16_t* q = p + (long)(row0 + (lane & 15)) * ld + kbase + ((lane >> 4) << 3);
  v8bf lo = *(const v8bf*)q;
  v8bf hi = *(const v8bf*)(q + 16);
  v16bf f;
#pragma unroll
  for (int i = 0; i < 8; ++i) { f[i] = lo[i]; f[i + 8] = hi[i]; }
  return f;
}

// ---------------------------------------------------------------------------
// GEMM: Y[M,N] = A[M,K] @ W[N,K]^T + bias[N]
// Block tile 128x128, BK=32, double-buffered async LDS staging, 8 waves:
// wave w -> m-quad = w>>1 (rows mq*32..mq*32+31, two 16-row WMMA tiles),
//           n-half = w&1  (64 cols = 4 WMMA tiles). 8 accumulators per wave.
// LDS rows padded to 40 bf16 (80B pitch) to spread banks.
// ---------------------------------------------------------------------------
constexpr int LDP  = 40;               // padded row pitch (bf16 elems)
constexpr int ASZ  = 128 * LDP;        // 5120 elems per A buffer
constexpr int BSZ  = 128 * LDP;        // 5120 elems per B buffer

__global__ __launch_bounds__(256) void gemm_bias_kernel(
    const bf16_t* __restrict__ A, const bf16_t* __restrict__ W,
    const float* __restrict__ bias, float* __restrict__ Y,
    int M, int N, int K) {
  __shared__ __align__(16) bf16_t smem[2][ASZ + BSZ];
  const int tid  = threadIdx.x;
  const int lane = tid & 31;
  const int wave = tid >> 5;
  const int mq   = wave >> 1;          // 0..3 (32-row group)
  const int nh   = wave & 1;           // 0..1 (64-col half)
  const int half = lane >> 4;
  const int m_blk = blockIdx.y << 7;
  const int n_blk = blockIdx.x << 7;

  // Staging: A and B are each 128 rows x 4 16B chunks = 512 chunks; 2/thread.
  auto stage = [&](int kk, int buf) {
#pragma unroll
    for (int t = 0; t < 2; ++t) {
      const int c = tid + (t << 8);
      const int row = c >> 2, sub = c & 3;
      cp16_g2l(A + (long)(m_blk + row) * K + kk + sub * 8,
               &smem[buf][row * LDP + sub * 8]);
      cp16_g2l(W + (long)(n_blk + row) * K + kk + sub * 8,
               &smem[buf][ASZ + row * LDP + sub * 8]);
    }
  };

  v8f acc[2][4] = {};
  stage(0, 0);
  wait_stage();
  __syncthreads();

  const int nk = K >> 5;
  for (int kt = 0; kt < nk; ++kt) {
    const int cur = kt & 1;
    if (kt + 1 < nk) stage((kt + 1) << 5, cur ^ 1);  // async-prefetch next

    // Preload ALL fragments first so WMMAs can issue back-to-back with
    // partial dscnt waits instead of a full drain per WMMA.
    v16bf afr[2];
#pragma unroll
    for (int s = 0; s < 2; ++s) {
      const bf16_t* ap =
          &smem[cur][(mq * 32 + s * 16 + (lane & 15)) * LDP + half * 8];
      v8bf alo = *(const v8bf*)ap;
      v8bf ahi = *(const v8bf*)(ap + 16);
#pragma unroll
      for (int i = 0; i < 8; ++i) { afr[s][i] = alo[i]; afr[s][i + 8] = ahi[i]; }
    }
    v16bf bfr[4];
#pragma unroll
    for (int j = 0; j < 4; ++j) {
      const bf16_t* bp =
          &smem[cur][ASZ + (nh * 64 + j * 16 + (lane & 15)) * LDP + half * 8];
      v8bf blo = *(const v8bf*)bp;
      v8bf bhi = *(const v8bf*)(bp + 16);
#pragma unroll
      for (int i = 0; i < 8; ++i) { bfr[j][i] = blo[i]; bfr[j][i + 8] = bhi[i]; }
    }
#pragma unroll
    for (int s = 0; s < 2; ++s)
#pragma unroll
      for (int j = 0; j < 4; ++j)
        acc[s][j] = __builtin_amdgcn_wmma_f32_16x16x32_bf16(
            false, afr[s], false, bfr[j], (short)0, acc[s][j], false, false);

    wait_stage();       // next-slice async staging landed in LDS
    __syncthreads();    // publish + protect cur buffer before re-stage
  }

  // C layout: VGPR g -> row +g (lanes 0-15) / +8+g (lanes 16-31), col = lane%16
#pragma unroll
  for (int s = 0; s < 2; ++s) {
    const int rbase = m_blk + mq * 32 + s * 16 + (half << 3);
#pragma unroll
    for (int j = 0; j < 4; ++j) {
      const int col = n_blk + nh * 64 + (j << 4) + (lane & 15);
      const float bv = bias[col];
#pragma unroll
      for (int g = 0; g < 8; ++g)
        Y[(long)(rbase + g) * N + col] = acc[s][j][g] + bv;
    }
  }
}

// ---------------------------------------------------------------------------
// Per-channel sum / sum-of-squares over all T*B rows (coalesced columns).
// grid = (E/256, rows/256). stats[0:E]=sum, stats[E:2E]=sumsq.
// ---------------------------------------------------------------------------
__global__ __launch_bounds__(256) void bn_stats_kernel(
    const float* __restrict__ Y, float* __restrict__ stats, int ncols) {
  const int col = blockIdx.x * 256 + threadIdx.x;
  const int r0 = blockIdx.y * 256;
  float s = 0.f, ss = 0.f;
  for (int r = 0; r < 256; ++r) {
    float v = Y[(long)(r0 + r) * ncols + col];
    s += v;
    ss += v * v;
  }
  atomicAdd(&stats[col], s);
  atomicAdd(&stats[ncols + col], ss);
}

// ---------------------------------------------------------------------------
// Fused BatchNorm + LIF (tau=2, vth=1, hard reset) over T, one thread per (b,e).
// mode 0: spikes time-first [t,b,e] bf16 -> q
// mode 1: spikes time-last  [b,e,t] bf16 -> k, v (t contiguous for WMMA K-dim)
// mode 2: x + spike [t,b,e] f32 -> final residual output
// ---------------------------------------------------------------------------
__global__ __launch_bounds__(256) void bn_lif_kernel(
    const float* __restrict__ Y, const float* __restrict__ stats,
    const float* __restrict__ gamma, const float* __restrict__ beta,
    const float* __restrict__ xin, bf16_t* __restrict__ out_tf,
    bf16_t* __restrict__ out_tl, float* __restrict__ out_res, int mode) {
  const long idx = (long)blockIdx.x * 256 + threadIdx.x; // b*E + e
  const int e = (int)(idx & (cE - 1));
  const float invN = 1.0f / (float)(cT * cB);
  const float mean = stats[e] * invN;
  const float var = stats[cE + e] * invN - mean * mean;
  const float sc = gamma[e] * rsqrtf(var + 1e-5f);
  const float sh = beta[e] - mean * sc;

  float v = 0.f;
  bf16_t loc[cT];
#pragma unroll
  for (int t = 0; t < cT; ++t) {
    const float y = Y[(long)t * cBE + idx] * sc + sh;
    v = 0.5f * (v + y);                         // v += (y - v)/tau, tau=2
    const float s = (v >= 1.0f) ? 1.0f : 0.0f;  // spike(v - vth)
    v *= (1.0f - s);                            // detached hard reset
    if (mode == 0)      out_tf[(long)t * cBE + idx] = (bf16_t)s;
    else if (mode == 1) loc[t] = (bf16_t)s;
    else                out_res[(long)t * cBE + idx] = xin[(long)t * cBE + idx] + s;
  }
  if (mode == 1) {
    v8bf* dst = (v8bf*)(out_tl + idx * cT);
    v8bf a, b2;
#pragma unroll
    for (int i = 0; i < 8; ++i) { a[i] = loc[i]; b2[i] = loc[8 + i]; }
    dst[0] = a;
    dst[1] = b2;
  }
}

// ---------------------------------------------------------------------------
// Fused linear attention per (b,h): one 128-thread block (4 waves).
// Stage 1: kv[d,e'] = sum_t k[t,d] v[t,e'] (K=16 zero-padded to 32; each wave
//          owns one 16-row d group, 4 WMMAs). kv stored transposed in LDS
//          (kvT[e'][d], +1 pad -> conflict-free).
// Stage 2: att[t,e'] = sum_d q[t,d] kv[d,e'] (K=64 -> 2 WMMAs per wave, B
//          fragments from LDS with f32->bf16 convert).
// ---------------------------------------------------------------------------
__global__ __launch_bounds__(128) void kv_att_kernel(
    const bf16_t* __restrict__ kTl, const bf16_t* __restrict__ vTl,
    const bf16_t* __restrict__ qS, bf16_t* __restrict__ attS) {
  __shared__ float kvT[cD * (cD + 1)];
  const int lane = threadIdx.x & 31;
  const int wave = threadIdx.x >> 5;
  const int b = blockIdx.x >> 4;
  const int h = blockIdx.x & 15;

  const bf16_t* kb_p = kTl + ((long)b * cE + h * cD) * cT; // [d][t], t contig
  const bf16_t* vb_p = vTl + ((long)b * cE + h * cD) * cT;

  // ---- Stage 1: kv = K^T V ------------------------------------------------
  const int m0 = wave << 4;
  v16bf afrag = {};        // elems 8..15 (t=16..31) stay zero
  {
    const bf16_t* q = kb_p + (long)(m0 + (lane & 15)) * cT + ((lane >> 4) << 3);
    v8bf lo = *(const v8bf*)q;
#pragma unroll
    for (int i = 0; i < 8; ++i) afrag[i] = lo[i];
  }
  v8f acc[4] = {};
#pragma unroll
  for (int j = 0; j < 4; ++j) {
    v16bf bfrag = {};      // elems 8..15 (t=16..31) stay zero
    {
      const bf16_t* q = vb_p + (long)((j << 4) + (lane & 15)) * cT + ((lane >> 4) << 3);
      v8bf lo = *(const v8bf*)q;
#pragma unroll
      for (int i = 0; i < 8; ++i) bfrag[i] = lo[i];
    }
    acc[j] = __builtin_amdgcn_wmma_f32_16x16x32_bf16(
        false, afrag, false, bfrag, (short)0, acc[j], false, false);
  }
#pragma unroll
  for (int j = 0; j < 4; ++j) {
    const int e2 = (j << 4) + (lane & 15);
    const int d0 = m0 + ((lane >> 4) << 3);
#pragma unroll
    for (int g = 0; g < 8; ++g)
      kvT[e2 * (cD + 1) + d0 + g] = acc[j][g];
  }
  __syncthreads();

  // ---- Stage 2: att = Q kv ------------------------------------------------
  const bf16_t* qb_p = qS + (long)b * cE + h * cD; // rows t, stride cBE
  const int e2 = (wave << 4) + (lane & 15);
  v8f catt = {};
#pragma unroll
  for (int kk = 0; kk < 2; ++kk) {
    const int kb2 = kk << 5;
    v16bf aq = loadA_g(qb_p, cBE, 0, kb2, lane);
    v16bf bkv;
    {
      const float* rowp = &kvT[e2 * (cD + 1) + kb2 + ((lane >> 4) << 3)];
#pragma unroll
      for (int i = 0; i < 16; ++i) bkv[i] = (bf16_t)rowp[i + (i >= 8 ? 8 : 0)];
    }
    catt = __builtin_amdgcn_wmma_f32_16x16x32_bf16(
        false, aq, false, bkv, (short)0, catt, false, false);
  }
  bf16_t* outp = attS + (long)b * cE + h * cD + e2;
  const int t0 = (lane >> 4) << 3;
#pragma unroll
  for (int g = 0; g < 8; ++g)
    outp[(long)(t0 + g) * cBE] = (bf16_t)catt[g];
}

// ---------------------------------------------------------------------------
__global__ void convert_bf16_kernel(const float* __restrict__ in,
                                    bf16_t* __restrict__ out, long n) {
  long i = (long)blockIdx.x * blockDim.x + threadIdx.x;
  if (i < n) out[i] = (bf16_t)in[i];
}

__global__ void zero_kernel(float* p, int n) {
  int i = blockIdx.x * blockDim.x + threadIdx.x;
  if (i < n) p[i] = 0.f;
}

// ---------------------------------------------------------------------------
extern "C" void kernel_launch(void* const* d_in, const int* in_sizes, int n_in,
                              void* d_out, int out_size, void* d_ws, size_t ws_size,
                              hipStream_t stream) {
  (void)in_sizes; (void)n_in; (void)out_size; (void)ws_size;
  const float* x = (const float*)d_in[0];
  const float* w_[4]  = {(const float*)d_in[1], (const float*)d_in[5],
                         (const float*)d_in[9], (const float*)d_in[13]};
  const float* bi_[4] = {(const float*)d_in[2], (const float*)d_in[6],
                         (const float*)d_in[10], (const float*)d_in[14]};
  const float* g_[4]  = {(const float*)d_in[3], (const float*)d_in[7],
                         (const float*)d_in[11], (const float*)d_in[15]};
  const float* be_[4] = {(const float*)d_in[4], (const float*)d_in[8],
                         (const float*)d_in[12], (const float*)d_in[16]};
  float* out = (float*)d_out;

  char* wp = (char*)d_ws;
  auto take = [&](size_t bytes) {
    char* p = wp;
    wp += (bytes + 255) & ~(size_t)255;
    return p;
  };
  bf16_t* xh = (bf16_t*)take((size_t)cTB * cE * sizeof(bf16_t));
  bf16_t* wh[4];
  for (int i = 0; i < 4; ++i)
    wh[i] = (bf16_t*)take((size_t)cE * cE * sizeof(bf16_t));
  float*  Y    = (float*)take((size_t)cTB * cE * sizeof(float));
  bf16_t* qS   = (bf16_t*)take((size_t)cTB * cE * sizeof(bf16_t)); // [t,b,e]
  bf16_t* kTl  = (bf16_t*)take((size_t)cTB * cE * sizeof(bf16_t)); // [b,e,t]
  bf16_t* vTl  = (bf16_t*)take((size_t)cTB * cE * sizeof(bf16_t)); // [b,e,t]
  bf16_t* attS = (bf16_t*)take((size_t)cTB * cE * sizeof(bf16_t)); // [t,b,e]
  float*  stats = (float*)take(2 * cE * sizeof(float));

  // 1) precision conversion
  convert_bf16_kernel<<<(int)(((long)cTB * cE) / 256), 256, 0, stream>>>(
      x, xh, (long)cTB * cE);
  for (int i = 0; i < 4; ++i)
    convert_bf16_kernel<<<(int)(((long)cE * cE) / 256), 256, 0, stream>>>(
        w_[i], wh[i], (long)cE * cE);

  const dim3 ggrid(cE / 128, cTB / 128); // GEMM grid (N/128, M/128)
  const dim3 sgrid(cE / 256, cTB / 256); // BN-stats grid

  // 2) q / k / v projections: GEMM -> BN stats -> fused BN+LIF
  const int modes[3] = {0, 1, 1};
  bf16_t* dst_tf[3] = {qS, nullptr, nullptr};
  bf16_t* dst_tl[3] = {nullptr, kTl, vTl};
  for (int p = 0; p < 3; ++p) {
    gemm_bias_kernel<<<ggrid, 256, 0, stream>>>(xh, wh[p], bi_[p], Y, cTB, cE, cE);
    zero_kernel<<<(2 * cE) / 256, 256, 0, stream>>>(stats, 2 * cE);
    bn_stats_kernel<<<sgrid, 256, 0, stream>>>(Y, stats, cE);
    bn_lif_kernel<<<(int)(cBE / 256), 256, 0, stream>>>(
        Y, stats, g_[p], be_[p], nullptr, dst_tf[p], dst_tl[p], nullptr, modes[p]);
  }

  // 3) fused kv + att per (b,h)
  kv_att_kernel<<<cB * cH, 128, 0, stream>>>(kTl, vTl, qS, attS);

  // 4) output projection + BN + LIF + residual
  gemm_bias_kernel<<<ggrid, 256, 0, stream>>>(attS, wh[3], bi_[3], Y, cTB, cE, cE);
  zero_kernel<<<(2 * cE) / 256, 256, 0, stream>>>(stats, 2 * cE);
  bn_stats_kernel<<<sgrid, 256, 0, stream>>>(Y, stats, cE);
  bn_lif_kernel<<<(int)(cBE / 256), 256, 0, stream>>>(
      Y, stats, g_[3], be_[3], x, nullptr, nullptr, out, 2);
}
